// ARERec_49589692400151
// MI455X (gfx1250) — matrix operand order repaired and checked by reading, not compiled, not run on hardware
//
#include <hip/hip_runtime.h>
#include <hip/hip_bf16.h>

typedef __attribute__((ext_vector_type(16))) __bf16 v16bf;
typedef __attribute__((ext_vector_type(8)))  float  v8f;
typedef int v4i_vec __attribute__((vector_size(16)));   // b128 async payload type

#define BATCH 512
#define SEQ   50
#define EMB   128
#define HEADS 8
#define RDIM  32
#define NCLS  5

// CDNA5 async global->LDS path (ASYNCcnt-tracked), guarded for portability.
#if defined(__has_builtin)
#  if __has_builtin(__builtin_amdgcn_global_load_async_to_lds_b128) && \
      __has_builtin(__builtin_amdgcn_s_wait_asynccnt)
#    define USE_ASYNC_LDS 1
#  endif
#endif
#ifndef USE_ASYNC_LDS
#  define USE_ASYNC_LDS 0
#endif

#define AS_GLOBAL __attribute__((address_space(1)))
#define AS_LOCAL  __attribute__((address_space(3)))

// ---------------------------------------------------------------------------
// Prep: convert a [128,128] f32 row-major weight into bf16 WMMA B-fragments.
// Layout: frag[ktile][ntile][lane][16] ; ktile in 0..3 (K step 32), ntile 0..7.
// Per ISA bf16 B 32x16: lanes 0-15 column N=lane hold K=k0..k0+7,k0+16..k0+23,
//                       lanes16-31 column N=lane-16 hold K=k0+8..k0+15,k0+24..k0+31.
// ---------------------------------------------------------------------------
__global__ void prep_wfrag(const float* __restrict__ W, __bf16* __restrict__ frag) {
    int t = blockIdx.x * blockDim.x + threadIdx.x;   // 1024 threads total
    if (t >= 4 * 8 * 32) return;
    int lane  = t & 31;
    int ntile = (t >> 5) & 7;
    int ktile = t >> 8;
    int ncol  = ntile * 16 + (lane & 15);
    int kb    = ktile * 32 + ((lane >= 16) ? 8 : 0);
    __bf16* dst = frag + ((size_t)ktile * 8 + ntile) * 512 + (size_t)lane * 16;
#pragma unroll
    for (int i = 0; i < 16; ++i)
        dst[i] = (__bf16)W[(size_t)(kb + (i < 8 ? i : 8 + i)) * EMB + ncol];
}

// ---------------------------------------------------------------------------
// WMMA GEMM: out[m, 0..127] = A[m, :] @ W + bias
//  mode 0: A row m = table[idx[m]]  (embedding gather, f32 -> bf16 on load)
//  mode 1: A row m = attn[b, e/16, s] * src[m, e]   (fused ctx = attn * v)
// Block = 256 threads = 8 waves; wave w owns M-tile rows [blk*128+w*16, +16),
// and iterates all 8 N-tiles. K loop: 4 steps of 32 (bf16 WMMA).
// ---------------------------------------------------------------------------
__global__ void gemm_bf16_wmma(int mode, int Mrows,
                               const float* __restrict__ src,
                               const int*   __restrict__ idx,
                               const float* __restrict__ attn,
                               const __bf16* __restrict__ wfrag,
                               const float* __restrict__ bias,
                               float* __restrict__ out) {
    int wave  = threadIdx.x >> 5;
    int lane  = threadIdx.x & 31;
    int mbase = blockIdx.x * 128 + wave * 16;
    if (mbase >= Mrows) return;
    int m = mbase + (lane & 15);

    const float* arow;
    if (mode == 0) arow = src + (size_t)idx[m] * EMB;
    else           arow = src + (size_t)m * EMB;

    int khalf = (lane < 16) ? 0 : 8;

    v8f acc[8];
#pragma unroll
    for (int nt = 0; nt < 8; ++nt) acc[nt] = (v8f){0,0,0,0,0,0,0,0};

    for (int k0 = 0; k0 < EMB; k0 += 32) {
        // ---- A fragment (16-bit A 16x32 layout) ----
        const float* p0 = arow + k0 + khalf;        // K = k0+khalf   .. +7
        const float* p1 = arow + k0 + 16 + khalf;   // K = k0+16+khalf.. +7
        float s0 = 1.0f, s1 = 1.0f;
        if (mode == 1) {
            int b = m / SEQ, s = m % SEQ;
            int h0 = (k0 + khalf) >> 4;             // chunk of 8 stays in one head
            int h1 = (k0 + 16 + khalf) >> 4;
            s0 = attn[((size_t)b * HEADS + h0) * SEQ + s];
            s1 = attn[((size_t)b * HEADS + h1) * SEQ + s];
        }
        v16bf a;
#pragma unroll
        for (int i = 0; i < 8; ++i) a[i]     = (__bf16)(p0[i] * s0);
#pragma unroll
        for (int i = 0; i < 8; ++i) a[i + 8] = (__bf16)(p1[i] * s1);

        const __bf16* wk0 = wfrag + (size_t)(k0 >> 5) * 4096 + (size_t)lane * 16;
#pragma unroll
        for (int nt = 0; nt < 8; ++nt) {
            v16bf bf = *(const v16bf*)(wk0 + (size_t)nt * 512);
            acc[nt] = __builtin_amdgcn_wmma_f32_16x16x32_bf16(
                false, a, false, bf, (short)0, acc[nt], false, false);
        }
    }

    // ---- store C (VGPR i -> row i + (lane>=16 ? 8:0), col = lane&15) ----
    int col0  = lane & 15;
    int rbase = mbase + ((lane < 16) ? 0 : 8);
#pragma unroll
    for (int nt = 0; nt < 8; ++nt) {
        int col = nt * 16 + col0;
        float bv = bias[col];
#pragma unroll
        for (int i = 0; i < 8; ++i)
            out[(size_t)(rbase + i) * EMB + col] = acc[nt][i] + bv;
    }
}

// ---------------------------------------------------------------------------
// Masked per-head softmax over S=50 neighbors. One block per batch row.
// logits[h,s] = (q[b,h*16:] . k[b,s,h*16:]) / 4  - 1e9*(neighbor<=0)
// ---------------------------------------------------------------------------
__global__ void attn_kernel(const float* __restrict__ q,        // [B,128]
                            const float* __restrict__ krow,     // [B*S,128]
                            const int*   __restrict__ neighbor, // [B*S]
                            float* __restrict__ attn) {         // [B,H,S]
    __shared__ float lg[HEADS * SEQ];
    int b = blockIdx.x;
    int t = threadIdx.x;                      // 256 threads
    for (int p = t; p < HEADS * SEQ; p += 256) {
        int h = p / SEQ, s = p % SEQ;
        const float* qp = q    + (size_t)b * EMB + h * 16;
        const float* kp = krow + ((size_t)b * SEQ + s) * EMB + h * 16;
        float d = 0.0f;
#pragma unroll
        for (int i = 0; i < 16; ++i) d += qp[i] * kp[i];
        d *= 0.25f;                           // 1/sqrt(16)
        if (neighbor[b * SEQ + s] <= 0) d -= 1e9f;
        lg[p] = d;
    }
    __syncthreads();
    if (t < HEADS) {
        float mx = -3.4e38f;
        for (int s = 0; s < SEQ; ++s) mx = fmaxf(mx, lg[t * SEQ + s]);
        float sum = 0.0f;
        for (int s = 0; s < SEQ; ++s) {
            float e = __expf(lg[t * SEQ + s] - mx);
            lg[t * SEQ + s] = e;
            sum += e;
        }
        float inv = 1.0f / sum;
        for (int s = 0; s < SEQ; ++s)
            attn[((size_t)b * HEADS + t) * SEQ + s] = lg[t * SEQ + s] * inv;
    }
}

// ---------------------------------------------------------------------------
// Fused region embedding: streams each 16KB LCU row exactly once.
// One block per batch row b; 256 threads: r = t>>3 (32 regions), sub = t&7
// (8 threads x 16 floats cover E=128). Running max over s kept in registers.
//
// CDNA5 data movement:
//  - nemb row for iteration s+1 staged into a double-buffered LDS slab via
//    GLOBAL_LOAD_ASYNC_TO_LDS_B128 (wave 0), drained with s_wait_asynccnt —
//    the fill is fully overlapped with iteration s's 32 KB LCU stream.
//  - next iteration's two 16 KB LCU rows prefetched with global_prefetch
//    (one 128B cacheline per thread: 256 threads x 128B = 32 KB exactly).
// ---------------------------------------------------------------------------
__global__ void region_kernel(const float* __restrict__ nemb,     // [B*S,128]
                              const float* __restrict__ item_tab, // [ITEMS,128]
                              const int*   __restrict__ item,     // [B]
                              const int*   __restrict__ neighbor, // [B*S]
                              const int*   __restrict__ seq,      // [B*S]
                              const float* __restrict__ Kui,      // [ALLSEQ,32,128]
                              const float* __restrict__ Kiu,      // [ALLSEQ,32,128]
                              float* __restrict__ rating) {       // [B,32]
    __shared__ float sh_n[2][EMB];
    __shared__ float sh_i[EMB];
    __shared__ int   sh_seq[SEQ];
    __shared__ float sh_w[SEQ];

    int b = blockIdx.x;
    int t = threadIdx.x;
    int r = t >> 3, sub = t & 7;

    if (t < SEQ) {
        sh_seq[t] = seq[b * SEQ + t];
        sh_w[t]   = (neighbor[b * SEQ + t] > 0) ? 1.0f : 0.0f;
    }
    if (t < EMB) sh_i[t] = item_tab[(size_t)item[b] * EMB + t];

#if USE_ASYNC_LDS
    if (t < 32) {   // prologue: stage row 0 into buffer 0 (16B per lane)
        const char* g = (const char*)(nemb + (size_t)b * SEQ * EMB) + t * 16;
        char*       l = (char*)&sh_n[0][0] + t * 16;
        __builtin_amdgcn_global_load_async_to_lds_b128(
            (AS_GLOBAL v4i_vec*)g, (AS_LOCAL v4i_vec*)l, 0, 0);
    }
#else
    if (t < EMB) sh_n[0][t] = nemb[(size_t)b * SEQ * EMB + t];
#endif
    __syncthreads();   // publish sh_seq / sh_w / sh_i

    float runmax = -3.4e38f;
    for (int s = 0; s < SEQ; ++s) {
        int buf = s & 1;
#if USE_ASYNC_LDS
        if (t < 32) {
            if (s + 1 < SEQ) {  // stage row s+1 into the other buffer
                const char* g =
                    (const char*)(nemb + ((size_t)b * SEQ + (s + 1)) * EMB) + t * 16;
                char* l = (char*)&sh_n[buf ^ 1][0] + t * 16;
                __builtin_amdgcn_global_load_async_to_lds_b128(
                    (AS_GLOBAL v4i_vec*)g, (AS_LOCAL v4i_vec*)l, 0, 0);
                __builtin_amdgcn_s_wait_asynccnt(1);   // row s has landed
            } else {
                __builtin_amdgcn_s_wait_asynccnt(0);
            }
        }
        __syncthreads();       // publish sh_n[buf] to all waves
#else
        __syncthreads();       // previous readers done before overwrite
        if (t < EMB) sh_n[buf][t] = nemb[((size_t)b * SEQ + s) * EMB + t];
        __syncthreads();
#endif
        // prefetch next iteration's LCU rows: one cacheline per thread
        if (s + 1 < SEQ) {
            size_t nr = (size_t)sh_seq[s + 1] * RDIM * EMB;
            const char* p = (t < 128)
                ? (const char*)(Kiu + nr) + (size_t)t * 128
                : (const char*)(Kui + nr) + (size_t)(t - 128) * 128;
            __builtin_prefetch(p, 0, 1);
        }

        size_t row = (size_t)sh_seq[s];
        const float* kiu = Kiu + (row * RDIM + r) * EMB + sub * 16;
        const float* kui = Kui + (row * RDIM + r) * EMB + sub * 16;
        float dn = 0.0f, di = 0.0f;
#pragma unroll
        for (int i = 0; i < 16; ++i) {
            dn += kiu[i] * sh_n[buf][sub * 16 + i];
            di += kui[i] * sh_i[sub * 16 + i];
        }
        // reduce over the 8 sub-lanes (wave32 shuffles)
        for (int msk = 4; msk >= 1; msk >>= 1) {
            dn += __shfl_xor(dn, msk, 32);
            di += __shfl_xor(di, msk, 32);
        }
        runmax = fmaxf(runmax, dn * di * sh_w[s]);

#if USE_ASYNC_LDS
        if (s + 1 < SEQ) __syncthreads();  // readers of sh_n[buf] done before
                                           // it is refilled at iteration s+1
#endif
    }
    if (sub == 0) rating[(size_t)b * RDIM + r] = runmax;
}

// ---------------------------------------------------------------------------
// Final tiny FC + softmax: one thread per batch row.
// ---------------------------------------------------------------------------
__global__ void fc_kernel(const float* __restrict__ rating, // [B,32]
                          const float* __restrict__ fc_w,   // [32,5]
                          const float* __restrict__ fc_b,   // [5]
                          float* __restrict__ out) {        // [B,5]
    int b = blockIdx.x * blockDim.x + threadIdx.x;
    if (b >= BATCH) return;
    float lg[NCLS];
#pragma unroll
    for (int c = 0; c < NCLS; ++c) lg[c] = fc_b[c];
    for (int r = 0; r < RDIM; ++r) {
        float v = rating[(size_t)b * RDIM + r];
#pragma unroll
        for (int c = 0; c < NCLS; ++c) lg[c] += v * fc_w[r * NCLS + c];
    }
    float mx = lg[0];
#pragma unroll
    for (int c = 1; c < NCLS; ++c) mx = fmaxf(mx, lg[c]);
    float sum = 0.0f;
#pragma unroll
    for (int c = 0; c < NCLS; ++c) { lg[c] = __expf(lg[c] - mx); sum += lg[c]; }
    float inv = 1.0f / sum;
#pragma unroll
    for (int c = 0; c < NCLS; ++c) out[(size_t)b * NCLS + c] = lg[c] * inv;
}

// ---------------------------------------------------------------------------
extern "C" void kernel_launch(void* const* d_in, const int* in_sizes, int n_in,
                              void* d_out, int out_size, void* d_ws, size_t ws_size,
                              hipStream_t stream) {
    (void)in_sizes; (void)n_in; (void)out_size; (void)ws_size;

    const int*   user     = (const int*)d_in[0];
    const int*   item     = (const int*)d_in[1];
    const int*   neighbor = (const int*)d_in[2];
    const int*   seq      = (const int*)d_in[3];
    const float* user_tab = (const float*)d_in[4];
    const float* item_tab = (const float*)d_in[5];
    const float* Kui      = (const float*)d_in[6];   // user_item_LCU
    const float* Kiu      = (const float*)d_in[7];   // item_user_LCU
    const float* wq = (const float*)d_in[8];  const float* bq = (const float*)d_in[9];
    const float* wk = (const float*)d_in[10]; const float* bk = (const float*)d_in[11];
    const float* wv = (const float*)d_in[12]; const float* bv = (const float*)d_in[13];
    const float* wo = (const float*)d_in[14]; const float* bo = (const float*)d_in[15];
    const float* fc_w = (const float*)d_in[16];
    const float* fc_b = (const float*)d_in[17];
    float* out = (float*)d_out;

    // ---- workspace carve-up (256B aligned) ----
    char* base = (char*)d_ws;
    size_t off = 0;
    auto carve = [&](size_t bytes) {
        char* p = base + off;
        off = (off + bytes + 255) & ~(size_t)255;
        return p;
    };
    const size_t FRAG_B = 4 * 8 * 32 * 16 * sizeof(__bf16);   // 32 KB per weight
    __bf16* fq = (__bf16*)carve(FRAG_B);
    __bf16* fk = (__bf16*)carve(FRAG_B);
    __bf16* fv = (__bf16*)carve(FRAG_B);
    __bf16* fo = (__bf16*)carve(FRAG_B);
    float* qbuf  = (float*)carve((size_t)BATCH * EMB * 4);            // 256 KB
    float* krow  = (float*)carve((size_t)BATCH * SEQ * EMB * 4);      // 12.5 MB
    float* vrow  = (float*)carve((size_t)BATCH * SEQ * EMB * 4);      // 12.5 MB
    float* attnb = (float*)carve((size_t)BATCH * HEADS * SEQ * 4);    // 800 KB
    float* nemb2 = krow;  // k rows are dead after attn_kernel: reuse for ctx@wo
    float* ratng = (float*)carve((size_t)BATCH * RDIM * 4);           // 64 KB

    const int MROWS = BATCH * SEQ;            // 25600, multiple of 128

    // 1) weight fragments (bf16, WMMA B layout)
    prep_wfrag<<<4, 256, 0, stream>>>(wq, fq);
    prep_wfrag<<<4, 256, 0, stream>>>(wk, fk);
    prep_wfrag<<<4, 256, 0, stream>>>(wv, fv);
    prep_wfrag<<<4, 256, 0, stream>>>(wo, fo);

    // 2) projections (gather fused into A-fragment load)
    gemm_bf16_wmma<<<BATCH / 128, 256, 0, stream>>>(0, BATCH, user_tab, user,
                                                    nullptr, fq, bq, qbuf);
    gemm_bf16_wmma<<<MROWS / 128, 256, 0, stream>>>(0, MROWS, user_tab, neighbor,
                                                    nullptr, fk, bk, krow);
    gemm_bf16_wmma<<<MROWS / 128, 256, 0, stream>>>(0, MROWS, user_tab, neighbor,
                                                    nullptr, fv, bv, vrow);

    // 3) masked per-head softmax
    attn_kernel<<<BATCH, 256, 0, stream>>>(qbuf, krow, neighbor, attnb);

    // 4) output projection with fused ctx = attn * v (A-side scaling)
    gemm_bf16_wmma<<<MROWS / 128, 256, 0, stream>>>(1, MROWS, vrow, nullptr,
                                                    attnb, fo, bo, nemb2);

    // 5) fused region embedding + masked max over S (memory-bound streamer)
    region_kernel<<<BATCH, 256, 0, stream>>>(nemb2, item_tab, item, neighbor,
                                             seq, Kui, Kiu, ratng);

    // 6) classifier softmax
    fc_kernel<<<(BATCH + 255) / 256, 256, 0, stream>>>(ratng, fc_w, fc_b, out);
}
